// GCNEncoder_57939108823254
// MI455X (gfx1250) — matrix operand order repaired and checked by reading, not compiled, not run on hardware
//
#include <hip/hip_runtime.h>

// ---------------------------------------------------------------------------
// GCN encoder for MI455X (gfx1250, wave32, WMMA).
//   x:[N,256] f32, edge_index:[2,E] i32, batch:[N] i32
//   3x (GEMM + edge scatter) with ReLU+LayerNorm between, mean-pool at end.
// GEMMs: v_wmma_f32_16x16x32_bf16. B slabs move global->LDS via CDNA5 async
// DMA (global_load_async_to_lds_b128, ASYNCcnt); B fragments double-buffered
// in registers (modulo-2 rotation, loads issued 2 WMMAs ahead).
// ---------------------------------------------------------------------------

#define NND   65536
#define NE    1048576
#define NGR   64
#define DIN   256
#define DHID  256
#define DOUT3 128
#define LNEPS 1e-5f

typedef __attribute__((ext_vector_type(16))) __bf16 v16bf;
typedef __attribute__((ext_vector_type(8)))  float  v8f;

__device__ __forceinline__ void atomAddF(float* p, float v) {
#if defined(__HIP_PLATFORM_AMD__)
    unsafeAtomicAdd(p, v);   // hardware global_atomic_add_f32
#else
    atomicAdd(p, v);
#endif
}

// 16B global -> LDS async copy (per-lane). VDST carries the LDS byte address
// (low 32 bits of the generic pointer = LDS offset per flat-aperture rules).
__device__ __forceinline__ void async_copy16(const void* gptr, void* lptr) {
    unsigned lds = (unsigned)(size_t)lptr;
    asm volatile("global_load_async_to_lds_b128 %0, %1, off"
                 :: "v"(lds), "v"(gptr) : "memory");
}

__device__ __forceinline__ void wait_async0() {
#if __has_builtin(__builtin_amdgcn_s_wait_asynccnt)
    __builtin_amdgcn_s_wait_asynccnt(0);
#else
    asm volatile("s_wait_asynccnt 0x0" ::: "memory");
#endif
}

// ---------------- degree / normalization ----------------
__global__ __launch_bounds__(256) void deg_count(const int* __restrict__ dst,
                                                 float* __restrict__ deg, int E) {
    int i = blockIdx.x * 256 + threadIdx.x;
    if (i < E) atomAddF(&deg[dst[i]], 1.0f);
}

__global__ __launch_bounds__(256) void deg_rsqrt(float* __restrict__ deg, int n) {
    int i = blockIdx.x * 256 + threadIdx.x;
    if (i < n) deg[i] = rsqrtf(deg[i] + 1.0f);   // +1 for self loop
}

__global__ __launch_bounds__(256) void count_nodes(const int* __restrict__ batch,
                                                   float* __restrict__ cnt, int n) {
    int i = blockIdx.x * 256 + threadIdx.x;
    if (i < n) atomAddF(&cnt[batch[i]], 1.0f);
}

// ---------------- f32 -> bf16 convert ----------------
__global__ __launch_bounds__(256) void cvt_bf16(const float* __restrict__ in,
                                                __bf16* __restrict__ out, int n) {
    int i = blockIdx.x * 256 + threadIdx.x;
    if (i < n) out[i] = (__bf16)in[i];
}

// ---------------- weight packing into WMMA B-fragment layout ----------------
// Slab-major so each kb-slab (NT fragments) is contiguous 16*NT*32 bf16 (NT*1024 bytes):
// Bp[((kb*NT + nt)*32 + lane)*16 + e] = W[k, n]
//   n = nt*16 + (lane&15), g = lane>>4: e<8 -> k = kb*32 + 8g + e,
//   e>=8 -> k = kb*32 + 16 + 8g + (e-8)   (matches A fragment k order)
__global__ __launch_bounds__(256) void pack_w(const float* __restrict__ W,
                                              __bf16* __restrict__ Bp,
                                              int K, int Dout) {
    int idx = blockIdx.x * 256 + threadIdx.x;
    if (idx >= K * Dout) return;
    int NT   = Dout >> 4;
    int e    = idx & 15;
    int lane = (idx >> 4) & 31;
    int blk  = idx >> 9;            // kb*NT + nt
    int nt   = blk % NT;
    int kb   = blk / NT;
    int g    = lane >> 4;
    int n    = nt * 16 + (lane & 15);
    int k    = kb * 32 + (e < 8 ? 8 * g + e : 16 + 8 * g + (e - 8));
    Bp[idx] = (__bf16)W[(size_t)k * Dout + n];
}

// ---------------- WMMA GEMM: Y[N,Dout] = A[N,K](bf16) * Bp(packed bf16) ----------------
// NTT = total 16-col tiles in Dout, NTW = tiles per wave (NTT/NTW waves share
// one 16-row m-tile). 8-wave blocks -> 8/(NTT/NTW) m-tiles per block.
// kb-slabs of B are double-buffered in LDS and filled with async global->LDS
// DMA issued before the WMMAs of the current slab; s_wait_asynccnt(0) +
// barrier closes each iteration. B fragments rotate through two independent
// register buffers, each reloaded in place 2 WMMAs before use.
// A 16x32 bf16 fragment lane layout (ISA 7.12.2): lane = m + 16g holds
//   e<8  -> k = 8g + e ; e>=8 -> k = 16 + 8g + (e-8)   (two 16B chunks/lane)
template <int NTT, int NTW>
__global__ __launch_bounds__(256) void gemm_wmma(const __bf16* __restrict__ A,
                                                 const __bf16* __restrict__ Bp,
                                                 float* __restrict__ Y, int K) {
    constexpr int Dout = NTT * 16;
    constexpr int SLB  = NTT * 1024; // bytes per kb-slab
    constexpr int PER  = NTT / 4;    // 16B chunks per thread per slab (256 thr)
    constexpr int WPM  = NTT / NTW;  // waves cooperating on one m-tile
    constexpr int MPB  = 8 / WPM;    // m-tiles per 8-wave block
    extern __shared__ char smem[];   // 2 * SLB bytes

    int wave = threadIdx.x >> 5;
    int lane = threadIdx.x & 31;
    int m0   = (blockIdx.x * MPB + (wave / WPM)) << 4;
    int half = wave % WPM;           // which Dout chunk this wave owns
    int g    = lane >> 4;
    int rc   = lane & 15;
    const int KB = K >> 5;

    v8f zero = {0.f,0.f,0.f,0.f,0.f,0.f,0.f,0.f};
    v8f acc[NTW];
#pragma unroll
    for (int i = 0; i < NTW; ++i) acc[i] = zero;

    const __bf16* arow = A + (size_t)(m0 + rc) * K;
    const char*   gBc  = (const char*)Bp;

    // prologue: slab 0 -> LDS buffer 0 via async DMA
#pragma unroll
    for (int i = 0; i < PER; ++i) {
        int off = (threadIdx.x + i * 256) * 16;
        async_copy16(gBc + off, smem + off);
    }
    wait_async0();
    __syncthreads();

    union afrag_t { v16bf v; uint4 u[2]; };

    for (int kb = 0; kb < KB; ++kb) {
        const int cur = kb & 1, nxt = cur ^ 1;
        // async DMA for next slab (runs while we do WMMAs on current slab)
        if (kb + 1 < KB) {
#pragma unroll
            for (int i = 0; i < PER; ++i) {
                int off = (threadIdx.x + i * 256) * 16;
                async_copy16(gBc + (size_t)(kb + 1) * SLB + off,
                             smem + nxt * SLB + off);
            }
        }
        if (kb + 2 < KB)
            __builtin_prefetch(arow + (size_t)(kb + 2) * 32, 0, 3);

        // A fragment (two contiguous 16B chunks per lane)
        afrag_t af;
        af.u[0] = *(const uint4*)(arow + kb * 32 + 8 * g);
        af.u[1] = *(const uint4*)(arow + kb * 32 + 16 + 8 * g);

        // B fragments: two independent register buffers, loads 2 WMMAs ahead
        const __bf16* sbase = (const __bf16*)(smem + cur * SLB)
                              + (size_t)half * NTW * 32 * 16;
        v16bf bf0 = *(const v16bf*)(sbase + (size_t)(0 * 32 + lane) * 16);
        v16bf bf1 = *(const v16bf*)(sbase + (size_t)(1 * 32 + lane) * 16);
#pragma unroll
        for (int nt = 0; nt < NTW; nt += 2) {
            acc[nt] = __builtin_amdgcn_wmma_f32_16x16x32_bf16(
                false, af.v, false, bf0, (short)0, acc[nt], false, false);
            if (nt + 2 < NTW)
                bf0 = *(const v16bf*)(sbase + (size_t)((nt + 2) * 32 + lane) * 16);
            acc[nt + 1] = __builtin_amdgcn_wmma_f32_16x16x32_bf16(
                false, af.v, false, bf1, (short)0, acc[nt + 1], false, false);
            if (nt + 3 < NTW)
                bf1 = *(const v16bf*)(sbase + (size_t)((nt + 3) * 32 + lane) * 16);
        }

        // close iteration: async writes visible, all waves done reading cur
        wait_async0();
        __syncthreads();
    }

    // C/D layout: VGPR r, lanes 0-15 -> M=r, lanes 16-31 -> M=r+8; N = lane&15
#pragma unroll
    for (int nt = 0; nt < NTW; ++nt)
#pragma unroll
        for (int r = 0; r < 8; ++r)
            Y[(size_t)(m0 + r + 8 * g) * Dout + (half * NTW + nt) * 16 + rc] = acc[nt][r];
}

// ---------------- edge scatter: scat[dst] += t[src] * dis[src]*dis[dst] ----------------
__global__ __launch_bounds__(256) void scatter_edges(const int* __restrict__ src,
                                                     const int* __restrict__ dst,
                                                     const float* __restrict__ dis,
                                                     const float* __restrict__ t,
                                                     float* __restrict__ scat,
                                                     int E, int D) {
    int wave = (blockIdx.x * 256 + threadIdx.x) >> 5;
    int lane = threadIdx.x & 31;
    if (wave >= E) return;
    int s = src[wave], d = dst[wave];
    float w = dis[s] * dis[d];
    const float4* ts4 = (const float4*)(t + (size_t)s * D);
    float* od = scat + (size_t)d * D;
    const int D4 = D >> 2;
    for (int f = lane; f < D4; f += 32) {
        float4 v = ts4[f];
        atomAddF(&od[f * 4 + 0], v.x * w);
        atomAddF(&od[f * 4 + 1], v.y * w);
        atomAddF(&od[f * 4 + 2], v.z * w);
        atomAddF(&od[f * 4 + 3], v.w * w);
    }
}

// ---------------- self-loop + bias + ReLU + LayerNorm -> bf16 (wave per node) ----------------
__global__ __launch_bounds__(256) void post_ln(const float* __restrict__ scat,
                                               const float* __restrict__ t,
                                               const float* __restrict__ dis,
                                               const float* __restrict__ b,
                                               const float* __restrict__ gam,
                                               const float* __restrict__ bet,
                                               __bf16* __restrict__ hbf) {
    const int D = 256;
    int node = (blockIdx.x * 256 + threadIdx.x) >> 5;
    int lane = threadIdx.x & 31;
    float sl = dis[node]; sl *= sl;
    float v[8], sum = 0.f, sq = 0.f;
#pragma unroll
    for (int i = 0; i < 8; ++i) {
        int f = lane + i * 32;
        float x = scat[(size_t)node * D + f] + t[(size_t)node * D + f] * sl + b[f];
        x = fmaxf(x, 0.f);
        v[i] = x; sum += x; sq += x * x;
    }
#pragma unroll
    for (int o = 16; o > 0; o >>= 1) {
        sum += __shfl_xor(sum, o, 32);
        sq  += __shfl_xor(sq,  o, 32);
    }
    float mu  = sum * (1.0f / D);
    float var = sq * (1.0f / D) - mu * mu;
    float inv = rsqrtf(var + LNEPS);
#pragma unroll
    for (int i = 0; i < 8; ++i) {
        int f = lane + i * 32;
        hbf[(size_t)node * D + f] = (__bf16)((v[i] - mu) * inv * gam[f] + bet[f]);
    }
}

// ---------------- final layer: h_out + pooled sums ----------------
__global__ __launch_bounds__(256) void final_out(const float* __restrict__ scat,
                                                 const float* __restrict__ t,
                                                 const float* __restrict__ dis,
                                                 const float* __restrict__ b3,
                                                 const int* __restrict__ batch,
                                                 float* __restrict__ h_out,
                                                 float* __restrict__ pool) {
    const int D = 128;
    int node = (blockIdx.x * 256 + threadIdx.x) >> 5;
    int lane = threadIdx.x & 31;
    float sl = dis[node]; sl *= sl;
    int grp = batch[node];
#pragma unroll
    for (int i = 0; i < 4; ++i) {
        int f = lane + i * 32;
        float x = scat[(size_t)node * D + f] + t[(size_t)node * D + f] * sl + b3[f];
        h_out[(size_t)node * D + f] = x;
        atomAddF(&pool[(size_t)grp * D + f], x);
    }
}

__global__ __launch_bounds__(256) void pool_div(float* __restrict__ pool,
                                                const float* __restrict__ cnt, int n) {
    int i = blockIdx.x * 256 + threadIdx.x;
    if (i < n) pool[i] /= fmaxf(cnt[i >> 7], 1.0f);
}

// ---------------------------------------------------------------------------
extern "C" void kernel_launch(void* const* d_in, const int* in_sizes, int n_in,
                              void* d_out, int out_size, void* d_ws, size_t ws_size,
                              hipStream_t stream) {
    const float* x    = (const float*)d_in[0];
    const int*   ei   = (const int*)d_in[1];          // [2,E] flat: src then dst
    const int*   bat  = (const int*)d_in[2];
    const float* W1 = (const float*)d_in[3];  const float* b1 = (const float*)d_in[4];
    const float* g1 = (const float*)d_in[5];  const float* be1= (const float*)d_in[6];
    const float* W2 = (const float*)d_in[7];  const float* b2 = (const float*)d_in[8];
    const float* g2 = (const float*)d_in[9];  const float* be2= (const float*)d_in[10];
    const float* W3 = (const float*)d_in[11]; const float* b3 = (const float*)d_in[12];
    const int* src = ei;
    const int* dst = ei + NE;

    float* h_out = (float*)d_out;                   // [N,128]
    float* pool  = h_out + (size_t)NND * DOUT3;     // [64,128]

    // workspace carve-up (256B aligned)
    char* w = (char*)d_ws;
    size_t off = 0;
    auto take = [&](size_t bytes) { void* p = w + off; off = (off + bytes + 255) & ~(size_t)255; return p; };
    float*  dis  = (float*) take((size_t)NND * 4);            // deg -> rsqrt(deg+1)
    float*  t    = (float*) take((size_t)NND * 256 * 4);      // GEMM output
    float*  scat = (float*) take((size_t)NND * 256 * 4);      // scatter accumulator
    __bf16* hbf  = (__bf16*)take((size_t)NND * 256 * 2);      // bf16 activations
    __bf16* wpk  = (__bf16*)take((size_t)256 * 256 * 2);      // packed weights
    float*  cnt  = (float*) take((size_t)NGR * 4);
    (void)ws_size; (void)in_sizes; (void)n_in; (void)out_size;

    // --- degree / dis / graph counts / pool init ---
    hipMemsetAsync(dis, 0, (size_t)NND * 4, stream);
    hipMemsetAsync(cnt, 0, (size_t)NGR * 4, stream);
    hipMemsetAsync(pool, 0, (size_t)NGR * DOUT3 * 4, stream);
    deg_count<<<NE / 256, 256, 0, stream>>>(dst, dis, NE);
    deg_rsqrt<<<NND / 256, 256, 0, stream>>>(dis, NND);
    count_nodes<<<NND / 256, 256, 0, stream>>>(bat, cnt, NND);

    // --- bf16 input activations ---
    cvt_bf16<<<(NND * DIN) / 256, 256, 0, stream>>>(x, hbf, NND * DIN);

    const int mblocks = NND / 16 / 4;       // 4 m-tiles per block (2 waves each)
    const int eblocks = NE / 8;             // wave per edge
    const int nblocks = NND / 8;            // wave per node
    const size_t lds16 = 2u * 16 * 1024;    // 32 KB (Dout=256)
    const size_t lds8  = 2u *  8 * 1024;    // 16 KB (Dout=128)

    // ---------------- layer 1: conv -> relu -> ln ----------------
    pack_w<<<(DIN * DHID) / 256, 256, 0, stream>>>(W1, wpk, DIN, DHID);
    gemm_wmma<16, 8><<<mblocks, 256, lds16, stream>>>(hbf, wpk, t, DIN);
    hipMemsetAsync(scat, 0, (size_t)NND * DHID * 4, stream);
    scatter_edges<<<eblocks, 256, 0, stream>>>(src, dst, dis, t, scat, NE, DHID);
    post_ln<<<nblocks, 256, 0, stream>>>(scat, t, dis, b1, g1, be1, hbf);

    // ---------------- layer 2: conv -> relu -> ln ----------------
    pack_w<<<(DHID * DHID) / 256, 256, 0, stream>>>(W2, wpk, DHID, DHID);
    gemm_wmma<16, 8><<<mblocks, 256, lds16, stream>>>(hbf, wpk, t, DHID);
    hipMemsetAsync(scat, 0, (size_t)NND * DHID * 4, stream);
    scatter_edges<<<eblocks, 256, 0, stream>>>(src, dst, dis, t, scat, NE, DHID);
    post_ln<<<nblocks, 256, 0, stream>>>(scat, t, dis, b2, g2, be2, hbf);

    // ---------------- layer 3: conv only + pooling ----------------
    pack_w<<<(DHID * DOUT3) / 256, 256, 0, stream>>>(W3, wpk, DHID, DOUT3);
    gemm_wmma<8, 4><<<mblocks, 256, lds8, stream>>>(hbf, wpk, t, DHID);
    hipMemsetAsync(scat, 0, (size_t)NND * DOUT3 * 4, stream);
    scatter_edges<<<eblocks, 256, 0, stream>>>(src, dst, dis, t, scat, NE, DOUT3);
    final_out<<<nblocks, 256, 0, stream>>>(scat, t, dis, b3, bat, h_out, pool);
    pool_div<<<(NGR * DOUT3) / 256, 256, 0, stream>>>(pool, cnt, NGR * DOUT3);
}
